// ConditionalSimNet2_64269890617915
// MI455X (gfx1250) — matrix operand order repaired and checked by reading, not compiled, not run on hardware
//
#include <hip/hip_runtime.h>
#include <hip/hip_bf16.h>

typedef __attribute__((ext_vector_type(16))) __bf16 v16bf;
typedef __attribute__((ext_vector_type(8)))  __bf16 v8bf;
typedef __attribute__((ext_vector_type(8)))  float  v8f;
typedef __attribute__((ext_vector_type(4)))  float  v4f;

#define B_ROWS   128
#define D_DIM    1024
#define FEAT_INN 2048
#define N_COND   66
#define OUT_ROWS 132
#define CAT2     24
#define A_PITCH  56   // bf16 elements per LDS row: 112B, 16B aligned, conflict-light

static __device__ __forceinline__ void bf16_split(float w, __bf16& hi, __bf16& lo) {
    __bf16 h = (__bf16)w;          // RNE fp32 -> bf16
    hi = h;
    lo = (__bf16)(w - (float)h);   // residual in bf16
}

// ---------------------------------------------------------------------------
// bf16x3 (split hi/lo) WMMA GEMM:  out[row, n, col] = A[row,:] @ W_n[:, col]
//   A: [128 x K] fp32, row stride lda
//   W: Wbase + n*K*1024, [K x 1024] fp32 (streamed once from HBM)
//   epilogue: (+ bias[n*biasNStride + col]) (* mask[n*1024 + col] if WITH_MASK)
// Block: 256 threads = 8 waves. Wave w owns output cols [blk*128 + w*16, +16),
// all 128 rows -> 8 accum tiles of 16x16 (v8f each). K loop in chunks of 32.
// W tile for iteration kc+32 is prefetched into registers while the WMMAs for
// kc run (software pipeline over the HBM-critical stream).
// ---------------------------------------------------------------------------
template <bool WITH_MASK>
__global__ __launch_bounds__(256)
void gemm_bf16x3(const float* __restrict__ A, int lda, int K,
                 const float* __restrict__ Wbase,
                 const float* __restrict__ bias, int biasNStride,
                 const float* __restrict__ mask,
                 float* __restrict__ out, size_t outRowStride, size_t outNStride)
{
    __shared__ __bf16 ldsAhi[128 * A_PITCH];
    __shared__ __bf16 ldsAlo[128 * A_PITCH];

    const int n     = blockIdx.y;
    const float* W  = Wbase + (size_t)n * (size_t)K * D_DIM;
    const int tid   = threadIdx.x;
    const int wave  = tid >> 5;
    const int lane  = tid & 31;
    const int lhalf = (lane >> 4) & 1;   // 0: lanes 0-15, 1: lanes 16-31
    const int l16   = lane & 15;
    const int col0  = blockIdx.x * 128 + wave * 16;

    v8f acc[8];
    #pragma unroll
    for (int m = 0; m < 8; ++m) {
        #pragma unroll
        for (int r = 0; r < 8; ++r) acc[m][r] = 0.0f;
    }

    // ---- W pipeline prologue: load tile for kc = 0 into registers ----------
    // B layout (32x16 bf16): lane col = l16, elems e=0..15 <-> K = kc + lhalf*16 + e
    const float* wcol = W + (col0 + l16);
    float wcur[16];
    #pragma unroll
    for (int e = 0; e < 16; ++e)
        wcur[e] = wcol[(size_t)(lhalf * 16 + e) * D_DIM];

    for (int kc = 0; kc < K; kc += 32) {
        // ---- cooperative stage of A[0:128][kc:kc+32] as bf16 hi/lo in LDS ----
        {
            const int arow  = tid >> 1;
            const int ahalf = (tid & 1) * 16;
            const float* ap = A + (size_t)arow * lda + kc + ahalf;
            v4f f0 = *(const v4f*)(ap);
            v4f f1 = *(const v4f*)(ap + 4);
            v4f f2 = *(const v4f*)(ap + 8);
            v4f f3 = *(const v4f*)(ap + 12);
            float f[16];
            #pragma unroll
            for (int e = 0; e < 4; ++e) { f[e] = f0[e]; f[4+e] = f1[e]; f[8+e] = f2[e]; f[12+e] = f3[e]; }
            v8bf h0, h1, l0, l1;
            #pragma unroll
            for (int e = 0; e < 8; ++e) {
                __bf16 hh, ll;
                bf16_split(f[e],     hh, ll); h0[e] = hh; l0[e] = ll;
                bf16_split(f[8 + e], hh, ll); h1[e] = hh; l1[e] = ll;
            }
            __bf16* dh = &ldsAhi[arow * A_PITCH + ahalf];
            __bf16* dl = &ldsAlo[arow * A_PITCH + ahalf];
            *reinterpret_cast<v8bf*>(dh)     = h0;
            *reinterpret_cast<v8bf*>(dh + 8) = h1;
            *reinterpret_cast<v8bf*>(dl)     = l0;
            *reinterpret_cast<v8bf*>(dl + 8) = l1;
        }

        // ---- issue next W tile loads now; consumed next iteration ----------
        float wnext[16];
        if (kc + 32 < K) {
            #pragma unroll
            for (int e = 0; e < 16; ++e)
                wnext[e] = wcol[(size_t)(kc + 32 + lhalf * 16 + e) * D_DIM];
        } else {
            #pragma unroll
            for (int e = 0; e < 16; ++e) wnext[e] = wcur[e];
        }

        // ---- convert current W regs into hi/lo B fragments (no mem wait) ---
        v16bf bhi, blo;
        #pragma unroll
        for (int e = 0; e < 16; ++e) {
            __bf16 hh, ll;
            bf16_split(wcur[e], hh, ll);
            bhi[e] = hh;
            blo[e] = ll;
        }

        __syncthreads();

        // ---- 8 M-subtiles x 3 WMMAs (hi*hi + hi*lo + lo*hi) ----
        #pragma unroll
        for (int m = 0; m < 8; ++m) {
            // A layout (16x32 bf16): row = m*16 + l16; elems 0-15 at K offset
            // lhalf*16 + e (matches B)
            const __bf16* ph = &ldsAhi[(m * 16 + l16) * A_PITCH];
            const __bf16* pl = &ldsAlo[(m * 16 + l16) * A_PITCH];
            const int c0 = lhalf * 16;
            const int c1 = lhalf * 16 + 8;
            v8bf ah0 = *reinterpret_cast<const v8bf*>(ph + c0);
            v8bf ah1 = *reinterpret_cast<const v8bf*>(ph + c1);
            v8bf al0 = *reinterpret_cast<const v8bf*>(pl + c0);
            v8bf al1 = *reinterpret_cast<const v8bf*>(pl + c1);
            v16bf ahi, alo;
            #pragma unroll
            for (int e = 0; e < 8; ++e) {
                ahi[e] = ah0[e]; ahi[8 + e] = ah1[e];
                alo[e] = al0[e]; alo[8 + e] = al1[e];
            }
            acc[m] = __builtin_amdgcn_wmma_f32_16x16x32_bf16(false, ahi, false, bhi, (short)0, acc[m], false, false);
            acc[m] = __builtin_amdgcn_wmma_f32_16x16x32_bf16(false, ahi, false, blo, (short)0, acc[m], false, false);
            acc[m] = __builtin_amdgcn_wmma_f32_16x16x32_bf16(false, alo, false, bhi, (short)0, acc[m], false, false);
        }
        __syncthreads();

        #pragma unroll
        for (int e = 0; e < 16; ++e) wcur[e] = wnext[e];
    }

    // ---- epilogue: D layout (16x16 f32): VGPR r -> M = r (lanes<16) / r+8 ----
    const int ocol = col0 + l16;
    const float bia = bias[(size_t)n * biasNStride + ocol];
    float msk = 1.0f;
    if (WITH_MASK) msk = mask[(size_t)n * D_DIM + ocol];
    #pragma unroll
    for (int m = 0; m < 8; ++m) {
        #pragma unroll
        for (int r = 0; r < 8; ++r) {
            const int orow = m * 16 + (lhalf ? r + 8 : r);
            float v = acc[m][r] + bia;
            if (WITH_MASK) v *= msk;
            out[(size_t)orow * outRowStride + (size_t)n * outNStride + ocol] = v;
        }
    }
}

// ---------------------------------------------------------------------------
// cate_net layer 1: h[p,n] = relu(cat_enc[p,:] @ W1[:,n] + b1[n])
// ---------------------------------------------------------------------------
__global__ void catnet_layer1(const float* __restrict__ enc,
                              const float* __restrict__ W1,
                              const float* __restrict__ b1,
                              float* __restrict__ h)
{
    const int p = blockIdx.x;
    const int nn = threadIdx.x;
    if (nn < N_COND) {
        float s = b1[nn];
        #pragma unroll
        for (int k = 0; k < CAT2; ++k) s += enc[p * CAT2 + k] * W1[k * N_COND + nn];
        h[p * N_COND + nn] = fmaxf(s, 0.0f);
    }
}

// ---------------------------------------------------------------------------
// cate_net layer 2 + softmax: att[p,:] = softmax(h[p,:] @ W2 + b2)
// ---------------------------------------------------------------------------
__global__ void catnet_layer2(const float* __restrict__ h,
                              const float* __restrict__ W2,
                              const float* __restrict__ b2,
                              float* __restrict__ att)
{
    __shared__ float vals[N_COND];
    const int p = blockIdx.x;
    const int m = threadIdx.x;
    float lg = 0.0f;
    if (m < N_COND) {
        lg = b2[m];
        for (int nn = 0; nn < N_COND; ++nn) lg += h[p * N_COND + nn] * W2[nn * N_COND + m];
        vals[m] = lg;
    }
    __syncthreads();
    float mx = -3.4e38f;
    for (int nn = 0; nn < N_COND; ++nn) mx = fmaxf(mx, vals[nn]);
    float sum = 0.0f;
    for (int nn = 0; nn < N_COND; ++nn) sum += __expf(vals[nn] - mx);
    if (m < N_COND) att[p * N_COND + m] = __expf(lg - mx) / sum;
}

// ---------------------------------------------------------------------------
// Combine (in-place over d_out's cond region) + feature_x broadcast.
// Each thread owns one (b,d) column: loads embed[b, 0..65, d] into registers,
// overwrites out[b, p, d] = sum_n att[p,n]*e[n], writes out[b, 66+j, d] = x[b,d].
// att indexed uniformly -> scalar loads.
// ---------------------------------------------------------------------------
__global__ __launch_bounds__(256)
void combine_kernel(const float* __restrict__ att,
                    const float* __restrict__ x,
                    float* __restrict__ out)
{
    const int g = blockIdx.x * 256 + threadIdx.x;   // 0 .. B*D-1
    const int b = g >> 10;
    const int d = g & (D_DIM - 1);
    float* ob = out + (size_t)b * OUT_ROWS * D_DIM + d;

    float e[N_COND];
    #pragma unroll
    for (int nn = 0; nn < N_COND; ++nn) e[nn] = ob[(size_t)nn * D_DIM];

    const float xv = x[(size_t)b * D_DIM + d];

    #pragma unroll 1
    for (int p = 0; p < N_COND; ++p) {
        float a = 0.0f;
        #pragma unroll
        for (int nn = 0; nn < N_COND; ++nn) a += att[p * N_COND + nn] * e[nn];
        ob[(size_t)p * D_DIM] = a;
    }
    #pragma unroll 1
    for (int j = 0; j < N_COND; ++j) ob[(size_t)(N_COND + j) * D_DIM] = xv;
}

// ---------------------------------------------------------------------------
extern "C" void kernel_launch(void* const* d_in, const int* in_sizes, int n_in,
                              void* d_out, int out_size, void* d_ws, size_t ws_size,
                              hipStream_t stream)
{
    (void)in_sizes; (void)n_in; (void)out_size; (void)ws_size;
    const float* image  = (const float*)d_in[0];
    const float* W_emb  = (const float*)d_in[1];
    const float* b_emb  = (const float*)d_in[2];
    const float* W_rep  = (const float*)d_in[3];
    const float* b_rep  = (const float*)d_in[4];
    const float* mask_t = (const float*)d_in[5];
    const float* W1     = (const float*)d_in[6];
    const float* b1     = (const float*)d_in[7];
    const float* W2     = (const float*)d_in[8];
    const float* b2     = (const float*)d_in[9];
    const float* catenc = (const float*)d_in[10];
    float* out = (float*)d_out;

    float* h_ws   = (float*)d_ws;                  // 66*66 floats
    float* att_ws = h_ws + N_COND * N_COND;        // 66*66 floats
    float* x_ws   = att_ws + N_COND * N_COND;      // 128*1024 floats

    // 1) attention weights (tiny)
    catnet_layer1<<<N_COND, 96, 0, stream>>>(catenc, W1, b1, h_ws);
    catnet_layer2<<<N_COND, 96, 0, stream>>>(h_ws, W2, b2, att_ws);

    // 2) x = image @ W_emb + b_emb   -> x_ws [128,1024]
    gemm_bf16x3<false><<<dim3(8, 1), 256, 0, stream>>>(
        image, FEAT_INN, FEAT_INN, W_emb, b_emb, 0, nullptr,
        x_ws, (size_t)D_DIM, (size_t)0);

    // 3) embed[b,n,:] = mask[n,:] * (x @ W_rep[n] + b_rep[n]) staged into
    //    d_out's cond region out[b, n, :]
    gemm_bf16x3<true><<<dim3(8, N_COND), 256, 0, stream>>>(
        x_ws, D_DIM, D_DIM, W_rep, b_rep, D_DIM, mask_t,
        out, (size_t)OUT_ROWS * D_DIM, (size_t)D_DIM);

    // 4) in-place attention combine + feature_x broadcast
    combine_kernel<<<(B_ROWS * D_DIM) / 256, 256, 0, stream>>>(att_ws, x_ws, out);
}
